// GRUcell_10101763080864
// MI455X (gfx1250) — matrix operand order repaired and checked by reading, not compiled, not run on hardware
//
#include <hip/hip_runtime.h>
#include <stdint.h>

#define B_SZ  512
#define S_SZ  256
#define NIN   128
#define DH    1024
#define DC    1152   // NIN + DH
#define NOUT  128
#define BN_EPS 1e-5f

// LDS tile: 64 rows x 32 bf16 cols, row pad +8 bf16 -> 20 u32 per row.
#define LROW   20
#define LBUF   (64 * LROW)   // u32 per buffer (triple buffered)

typedef __attribute__((ext_vector_type(16))) __bf16 v16bf;
typedef __attribute__((ext_vector_type(8)))  float  v8f;

union __attribute__((aligned(16))) FragU {
    uint32_t u[8];
    v16bf    v;
};

__device__ __forceinline__ uint16_t f2bf(float f) {
    uint32_t u = __float_as_uint(f);
    return (uint16_t)((u + 0x7FFFu + ((u >> 16) & 1u)) >> 16);  // RNE
}

__device__ __forceinline__ float sigmoidf_(float x) {
    return 1.0f / (1.0f + expf(-x));
}

// Generic pointers to __shared__ carry the wave-relative LDS byte offset in
// their low 32 bits (ISA 10.2).
__device__ __forceinline__ uint32_t lds_addr(const uint32_t* p) {
    return (uint32_t)(uintptr_t)p;
}

// Async DMA one 16-byte chunk global -> LDS (ASYNCcnt-tracked).
__device__ __forceinline__ void async_copy_b128(uint32_t ldsoff, const void* g) {
    asm volatile("global_load_async_to_lds_b128 %0, %1, off"
                 :: "v"(ldsoff), "v"(g) : "memory");
}

// Async loads complete in order: waiting ASYNCcnt<=1 guarantees the older of
// two outstanding tile-DMAs has fully landed in LDS.
__device__ __forceinline__ void wait_async_le0() {
    asm volatile("s_wait_asynccnt 0x0" ::: "memory");
}
__device__ __forceinline__ void wait_async_le1() {
    asm volatile("s_wait_asynccnt 0x1" ::: "memory");
}

// ---------------------------------------------------------------------------
// Stage one 64-row x 32-col bf16 A-tile into LDS buffer `bufbase`.
//   X phase (kb*32 < ncolsX): load f32 pairs, convert to bf16, ds_store.
//   H phase: pure bf16 copy -> one async b128 per thread (256 threads = tile).
// Uniform branch (kb, ncolsX are wave-uniform).
// ---------------------------------------------------------------------------
__device__ __forceinline__ void stage_tile(int kb, int bufbase, uint32_t* ldsA,
                                           const float* __restrict__ Xsrc, int strideX,
                                           int ncolsX,
                                           const uint16_t* __restrict__ Hsrc, int strideH,
                                           int mblock, int tid) {
    if (kb * 32 < ncolsX) {
        const int cu  = tid & 15;          // u32 column -> bf16 cols 2cu, 2cu+1
        const int r0  = tid >> 4;          // rows r0, r0+16, r0+32, r0+48
        const int col = kb * 32 + cu * 2;
        #pragma unroll
        for (int i = 0; i < 4; ++i) {
            const int rr = r0 + i * 16;
            const float* p = Xsrc + (mblock + rr) * strideX + col;
            ldsA[bufbase + rr * LROW + cu] =
                (uint32_t)f2bf(p[0]) | ((uint32_t)f2bf(p[1]) << 16);
        }
    } else {
        const int row = tid >> 2;          // 0..63
        const int chk = tid & 3;           // 16B chunk within 64B row
        const uint16_t* g = Hsrc + (mblock + row) * strideH
                            + (kb * 32 - ncolsX) + chk * 8;
        async_copy_b128(lds_addr(ldsA + bufbase + row * LROW + chk * 4), g);
    }
}

// ---------------------------------------------------------------------------
// WMMA main loop, G weight matrices sharing one A stream.
// Block = 256 threads = 8 waves, wm(0..1) x wn(0..3); wave tile = 32x16
// (two 16x16 WMMA tiles per gate sharing one B fragment). Block tile 64x64.
// LDS triple-buffered: two tile-DMAs in flight, one barrier per K-step.
// B fragments come from Wq packed layout [kb][n][hv][v] -> 2x b128 per frag.
// ---------------------------------------------------------------------------
template <int G>
__device__ __forceinline__ void wmma_mainloop(
    const float* __restrict__ Xsrc, int strideX, int ncolsX,
    const uint16_t* __restrict__ Hsrc, int strideH,
    const uint32_t* __restrict__ Wq0, const uint32_t* __restrict__ Wq1,
    int Nld, int K, int mblock, int nblock, uint32_t* ldsA,
    v8f acc[][2]) {
    const int tid  = threadIdx.x;
    const int lane = tid & 31;
    const int wave = tid >> 5;
    const int wm   = wave >> 2;       // 0..1 : 32-row strip
    const int wn   = wave & 3;        // 0..3 : 16-col strip
    const int l15  = lane & 15;
    const int hv   = lane >> 4;
    const int n    = nblock + wn * 16 + l15;

    const int kbTot = K / 32;
    const int kbX   = ncolsX / 32;

    const uint32_t* wq[2] = {Wq0, Wq1};
    const int bOff = (n * 2 + hv) << 3;      // u32 offset within a kb slab

    v8f z = {};
    #pragma unroll
    for (int g = 0; g < G; ++g) { acc[g][0] = z; acc[g][1] = z; }

    stage_tile(0, 0, ldsA, Xsrc, strideX, ncolsX, Hsrc, strideH, mblock, tid);
    stage_tile(1, LBUF, ldsA, Xsrc, strideX, ncolsX, Hsrc, strideH, mblock, tid);

    for (int kb = 0; kb < kbTot; ++kb) {
        if (kb >= kbX) {
            if (kb + 1 < kbTot) wait_async_le1();   // current tile landed
            else                wait_async_le0();   // nothing younger in flight
        }
        __syncthreads();                 // tile kb visible; buffer kb+2 free
        if (kb + 2 < kbTot)
            stage_tile(kb + 2, ((kb + 2) % 3) * LBUF, ldsA,
                       Xsrc, strideX, ncolsX, Hsrc, strideH, mblock, tid);

        const int cur = (kb % 3) * LBUF;

        // A fragments (16x32 bf16 each): VGPR0-7 = K 16hv+0..15
        FragU a0, a1;
        const uint32_t* rp0 = ldsA + cur + (wm * 32 + l15) * LROW;
        const uint32_t* rp1 = rp0 + 16 * LROW;
        *(uint4*)&a0.u[0] = *(const uint4*)(rp0 + 8 * hv);
        *(uint4*)&a0.u[4] = *(const uint4*)(rp0 + 8 * hv + 4);
        *(uint4*)&a1.u[0] = *(const uint4*)(rp1 + 8 * hv);
        *(uint4*)&a1.u[4] = *(const uint4*)(rp1 + 8 * hv + 4);

        #pragma unroll
        for (int g = 0; g < G; ++g) {
            FragU b;  // 32x16 bf16: lanes 0-15 K 0..15, lanes 16-31 K 16..31
            const uint32_t* bp = wq[g] + ((kb * Nld) << 4) + bOff;
            *(uint4*)&b.u[0] = *(const uint4*)(bp);
            *(uint4*)&b.u[4] = *(const uint4*)(bp + 4);
            acc[g][0] = __builtin_amdgcn_wmma_f32_16x16x32_bf16(
                false, a0.v, false, b.v, (short)0, acc[g][0], false, false);
            acc[g][1] = __builtin_amdgcn_wmma_f32_16x16x32_bf16(
                false, a1.v, false, b.v, (short)0, acc[g][1], false, false);
        }
    }
}

// ---------------------------------------------------------------------------
// Gate GEMM (G gates sharing one A stream): pre_g = [x_t | h'] @ W_g^T with
// BN batch sum / sum^2 harvested from accumulators (one float atomic per
// output column per wave strip). Gate biases dropped: BN(x+b) == BN(x).
// ---------------------------------------------------------------------------
template <int G>
__global__ __launch_bounds__(256) void gemm_gate_kernel(
    const float* __restrict__ Xsrc, int strideX, int ncolsX,
    const uint16_t* __restrict__ Hsrc, int strideH,
    const uint32_t* __restrict__ Wq0, const uint32_t* __restrict__ Wq1, int K,
    float* __restrict__ pre0, float* __restrict__ pre1,
    float* __restrict__ sum0, float* __restrict__ sumsq0,
    float* __restrict__ sum1, float* __restrict__ sumsq1) {
    __shared__ uint32_t ldsA[3 * LBUF];
    const int mblock = blockIdx.x * 64;
    const int nblock = blockIdx.y * 64;

    v8f acc[G][2];
    wmma_mainloop<G>(Xsrc, strideX, ncolsX, Hsrc, strideH, Wq0, Wq1, DH, K,
                     mblock, nblock, ldsA, acc);

    const int tid  = threadIdx.x;
    const int lane = tid & 31;
    const int wave = tid >> 5;
    const int wm   = wave >> 2;
    const int wn   = wave & 3;
    const int l15  = lane & 15;
    const int hv   = lane >> 4;
    const int n    = nblock + wn * 16 + l15;
    const int m0   = mblock + wm * 32 + hv * 8;   // tile0 rows; tile1 = +16

    float* pres[2] = {pre0, pre1};
    float* sums[2] = {sum0, sum1};
    float* sqs [2] = {sumsq0, sumsq1};

    #pragma unroll
    for (int g = 0; g < G; ++g) {
        float s = 0.0f, q = 0.0f;
        #pragma unroll
        for (int v = 0; v < 8; ++v) {
            float v0 = acc[g][0][v];
            float v1 = acc[g][1][v];
            pres[g][(m0 + v) * DH + n]      = v0;
            pres[g][(m0 + 16 + v) * DH + n] = v1;
            s += v0 + v1;
            q += v0 * v0 + v1 * v1;
        }
        s += __shfl_xor(s, 16);
        q += __shfl_xor(q, 16);
        if (hv == 0) {
            atomicAdd(&sums[g][n], s);
            atomicAdd(&sqs[g][n], q);
        }
    }
}

// ---------------------------------------------------------------------------
// Output GEMM: out[:, t, :] = tanh(h_new @ W_o^T + b_o)
// ---------------------------------------------------------------------------
__global__ __launch_bounds__(256) void gemm_out_kernel(
    const uint16_t* __restrict__ hbf, const uint32_t* __restrict__ WqO,
    const float* __restrict__ b_o, float* __restrict__ out, int t) {
    __shared__ uint32_t ldsA[3 * LBUF];
    const int mblock = blockIdx.x * 64;
    const int nblock = blockIdx.y * 64;

    v8f acc[1][2];
    wmma_mainloop<1>(nullptr, 0, 0, hbf, DH, WqO, WqO, NOUT, DH,
                     mblock, nblock, ldsA, acc);

    const int tid  = threadIdx.x;
    const int lane = tid & 31;
    const int wave = tid >> 5;
    const int wm   = wave >> 2;
    const int wn   = wave & 3;
    const int l15  = lane & 15;
    const int hv   = lane >> 4;
    const int n    = nblock + wn * 16 + l15;
    const int m0   = mblock + wm * 32 + hv * 8;

    const float bias = b_o[n];
    #pragma unroll
    for (int v = 0; v < 8; ++v) {
        out[(m0 + v) * (S_SZ * NOUT) + t * NOUT + n]      = tanhf(acc[0][0][v] + bias);
        out[(m0 + 16 + v) * (S_SZ * NOUT) + t * NOUT + n] = tanhf(acc[0][1][v] + bias);
    }
}

// ---------------------------------------------------------------------------
// BN + sigmoid for u and r gates; emits u (f32) and bf16(r*h).
// ---------------------------------------------------------------------------
__global__ __launch_bounds__(256) void bnact_ur_kernel(
    const float* __restrict__ preU, const float* __restrict__ preR,
    const float* __restrict__ sum_u, const float* __restrict__ sumsq_u,
    const float* __restrict__ sum_r, const float* __restrict__ sumsq_r,
    const float* __restrict__ g_u, const float* __restrict__ be_u,
    const float* __restrict__ g_r, const float* __restrict__ be_r,
    const float* __restrict__ h, float* __restrict__ ubuf,
    uint16_t* __restrict__ rhbf) {
    const int idx = blockIdx.x * blockDim.x + threadIdx.x;
    const int f   = idx & (DH - 1);
    const float inv = 1.0f / (float)B_SZ;

    float mu = sum_u[f] * inv;
    float vu = sumsq_u[f] * inv - mu * mu;
    float su = g_u[f] * rsqrtf(vu + BN_EPS);
    float u  = sigmoidf_((preU[idx] - mu) * su + be_u[f]);

    float mr = sum_r[f] * inv;
    float vr = sumsq_r[f] * inv - mr * mr;
    float sr = g_r[f] * rsqrtf(vr + BN_EPS);
    float r  = sigmoidf_((preR[idx] - mr) * sr + be_r[f]);

    ubuf[idx] = u;
    rhbf[idx] = f2bf(r * h[idx]);
}

// ---------------------------------------------------------------------------
// BN + tanh for candidate, then h_new = (1-u)*c + u*h (in-place h update).
// ---------------------------------------------------------------------------
__global__ __launch_bounds__(256) void bnact_h_kernel(
    const float* __restrict__ preC,
    const float* __restrict__ sum_c, const float* __restrict__ sumsq_c,
    const float* __restrict__ g_h, const float* __restrict__ be_h,
    const float* __restrict__ ubuf, float* __restrict__ h,
    uint16_t* __restrict__ hbf) {
    const int idx = blockIdx.x * blockDim.x + threadIdx.x;
    const int f   = idx & (DH - 1);
    const float inv = 1.0f / (float)B_SZ;

    float mc = sum_c[f] * inv;
    float vc = sumsq_c[f] * inv - mc * mc;
    float sc = g_h[f] * rsqrtf(vc + BN_EPS);
    float c  = tanhf((preC[idx] - mc) * sc + be_h[f]);

    float u  = ubuf[idx];
    float hn = (1.0f - u) * c + u * h[idx];
    h[idx]   = hn;
    hbf[idx] = f2bf(hn);
}

// ---------------------------------------------------------------------------
// Utility kernels
// ---------------------------------------------------------------------------
__global__ void zero_f32_kernel(float* __restrict__ p, int n) {
    const int i = blockIdx.x * blockDim.x + threadIdx.x;
    if (i < n) p[i] = 0.0f;
}

__global__ void init_h_kernel(float* __restrict__ h, uint16_t* __restrict__ hbf) {
    const int i = blockIdx.x * blockDim.x + threadIdx.x;
    h[i] = 0.0f;
    hbf[i] = 0;
}

// Pack W (N rows x K cols, f32 row-major) into WMMA-B-fragment order:
//   Wq[kb][n][hv][v] (v innermost, u32 = bf16 pair for K = kb*32 + hv*16 + 2v)
// so one B fragment = two contiguous b128 loads per lane.
__global__ void pack_w_kernel(const float* __restrict__ W, uint32_t* __restrict__ out,
                              int N, int K) {
    const int idx = blockIdx.x * blockDim.x + threadIdx.x;
    if (idx >= N * (K / 2)) return;
    const int k2 = idx / N;
    const int n  = idx % N;
    const int kb = k2 >> 4;
    const int r  = k2 & 15;
    const int hv = r >> 3;
    const int vv = r & 7;
    const float* p = W + (long)n * K + 2 * k2;
    out[(((kb * N + n) * 2 + hv) << 3) + vv] =
        (uint32_t)f2bf(p[0]) | ((uint32_t)f2bf(p[1]) << 16);
}

// ---------------------------------------------------------------------------
extern "C" void kernel_launch(void* const* d_in, const int* in_sizes, int n_in,
                              void* d_out, int out_size, void* d_ws, size_t ws_size,
                              hipStream_t stream) {
    (void)in_sizes; (void)n_in; (void)out_size; (void)ws_size;

    const float* X    = (const float*)d_in[0];
    const float* W_u  = (const float*)d_in[1];
    const float* W_r  = (const float*)d_in[3];
    const float* W_h  = (const float*)d_in[5];
    const float* W_o  = (const float*)d_in[7];
    const float* b_o  = (const float*)d_in[8];
    const float* g_u  = (const float*)d_in[9];
    const float* be_u = (const float*)d_in[10];
    const float* g_r  = (const float*)d_in[11];
    const float* be_r = (const float*)d_in[12];
    const float* g_h  = (const float*)d_in[13];
    const float* be_h = (const float*)d_in[14];
    float* out = (float*)d_out;

    // ---- workspace carve (all offsets 256B aligned) ----
    uint8_t* w = (uint8_t*)d_ws;
    auto alloc = [&](size_t bytes) -> uint8_t* {
        uint8_t* p = w;
        w += (bytes + 255) & ~(size_t)255;
        return p;
    };
    uint32_t* WqU  = (uint32_t*)alloc((size_t)(DC / 2) * DH * 4);
    uint32_t* WqR  = (uint32_t*)alloc((size_t)(DC / 2) * DH * 4);
    uint32_t* WqH  = (uint32_t*)alloc((size_t)(DC / 2) * DH * 4);
    uint32_t* WqO  = (uint32_t*)alloc((size_t)(DH / 2) * NOUT * 4);
    float*    hbuf = (float*)   alloc((size_t)B_SZ * DH * 4);
    uint16_t* hbf  = (uint16_t*)alloc((size_t)B_SZ * DH * 2);
    uint16_t* rhbf = (uint16_t*)alloc((size_t)B_SZ * DH * 2);
    float*    ubuf = (float*)   alloc((size_t)B_SZ * DH * 4);
    float*    preU = (float*)   alloc((size_t)B_SZ * DH * 4);
    float*    preR = (float*)   alloc((size_t)B_SZ * DH * 4);
    float*    preC = (float*)   alloc((size_t)B_SZ * DH * 4);
    float*    stats = (float*)  alloc((size_t)6 * DH * 4);
    float* sum_u = stats + 0 * DH;  float* sumsq_u = stats + 1 * DH;
    float* sum_r = stats + 2 * DH;  float* sumsq_r = stats + 3 * DH;
    float* sum_c = stats + 4 * DH;  float* sumsq_c = stats + 5 * DH;

    // ---- one-time per launch: pack weights, init h ----
    {
        int n = DH * (DC / 2);
        pack_w_kernel<<<(n + 255) / 256, 256, 0, stream>>>(W_u, WqU, DH, DC);
        pack_w_kernel<<<(n + 255) / 256, 256, 0, stream>>>(W_r, WqR, DH, DC);
        pack_w_kernel<<<(n + 255) / 256, 256, 0, stream>>>(W_h, WqH, DH, DC);
        int no = NOUT * (DH / 2);
        pack_w_kernel<<<(no + 255) / 256, 256, 0, stream>>>(W_o, WqO, NOUT, DH);
        init_h_kernel<<<(B_SZ * DH) / 256, 256, 0, stream>>>(hbuf, hbf);
    }

    const dim3 gGate(B_SZ / 64, DH / 64);   // 8 x 16 = 128 blocks
    const dim3 gOut (B_SZ / 64, NOUT / 64); // 8 x 2  = 16 blocks
    const dim3 blk(256);
    const int  nBN = (B_SZ * DH) / 256;     // 2048

    for (int t = 0; t < S_SZ; ++t) {
        const float* Xt = X + (long)t * NIN;  // row stride S*NIN

        zero_f32_kernel<<<(6 * DH) / 256, 256, 0, stream>>>(stats, 6 * DH);

        // u and r gates fused: one A stream, two weight matrices
        gemm_gate_kernel<2><<<gGate, blk, 0, stream>>>(
            Xt, S_SZ * NIN, NIN, hbf, DH, WqU, WqR, DC,
            preU, preR, sum_u, sumsq_u, sum_r, sumsq_r);

        bnact_ur_kernel<<<nBN, 256, 0, stream>>>(preU, preR, sum_u, sumsq_u,
                                                 sum_r, sumsq_r, g_u, be_u, g_r, be_r,
                                                 hbuf, ubuf, rhbf);

        gemm_gate_kernel<1><<<gGate, blk, 0, stream>>>(
            Xt, S_SZ * NIN, NIN, rhbf, DH, WqH, WqH, DC,
            preC, preC, sum_c, sumsq_c, sum_c, sumsq_c);

        bnact_h_kernel<<<nBN, 256, 0, stream>>>(preC, sum_c, sumsq_c, g_h, be_h,
                                                ubuf, hbuf, hbf);

        gemm_out_kernel<<<gOut, blk, 0, stream>>>(hbf, WqO, b_o, out, t);
    }
}
